// canODE_transformer_16518444221244
// MI455X (gfx1250) — compile-verified, matrix operand order checked by reading, not run on hardware
//
#include <hip/hip_runtime.h>
#include <hip/hip_bf16.h>
#include <math.h>

// Problem constants (match reference)
#define NB     8          // batch
#define DDIM   10000      // data dim
#define KC     256        // condensed length
#define DM     512        // d_model
#define NH     8          // heads
#define NL     6          // layers
#define FF     2048       // ffn dim
#define DH     64         // head dim
#define NT     (NB*KC)    // tokens = 2048
#define BH     (NB*NH)    // 64 attention matrices
#define NSTEPS 2

typedef _Float16 half8  __attribute__((ext_vector_type(8)));
typedef _Float16 half16 __attribute__((ext_vector_type(16)));
typedef float    float8 __attribute__((ext_vector_type(8)));

// ---------------------------------------------------------------------------
// WMMA helpers: NT GEMM fragments for V_WMMA_F32_16X16X32_F16.
// A (16x32, f16): lane m = lane&15, group g = lane>>4 holds K halves
//   [kk + g*8 .. +7] in VGPR 0-3 and [kk + 16 + g*8 .. +7] in VGPR 4-7:
//   two contiguous 16B chunks of the row -> two b128 loads.
// B (32x16): same pattern with lane = n over the (row-major [N,K]) weight.
// ---------------------------------------------------------------------------
__device__ __forceinline__ half16 load_frag16(const _Float16* __restrict__ p, int g) {
  const half8 lo = *(const half8*)(p + g * 8);
  const half8 hi = *(const half8*)(p + 16 + g * 8);
  half16 f;
#pragma unroll
  for (int i = 0; i < 8; ++i) { f[i] = lo[i]; f[8 + i] = hi[i]; }
  return f;
}

// One wave computes a 32(M) x 64(N) tile: 2 A-frags, each B-frag reused twice.
// acc[u][t]: u = 16-row subtile (0..1), t = 16-col subtile (0..3).
__device__ __forceinline__ void wmma_nt_32x64(const _Float16* __restrict__ Abase, int lda,
                                              const _Float16* __restrict__ Bbase, int ldb,
                                              int Kd, int lane, float8 acc[2][4]) {
  const int g = lane >> 4;
  const int r = lane & 15;
  const _Float16* arow0 = Abase + (size_t)r * lda;
  const _Float16* arow1 = Abase + (size_t)(16 + r) * lda;
  for (int kk = 0; kk < Kd; kk += 32) {
    half16 a0 = load_frag16(arow0 + kk, g);
    half16 a1 = load_frag16(arow1 + kk, g);
#pragma unroll
    for (int t = 0; t < 4; ++t) {
      half16 b = load_frag16(Bbase + (size_t)(t * 16 + r) * ldb + kk, g);
      acc[0][t] = __builtin_amdgcn_wmma_f32_16x16x32_f16(
          false, a0, false, b, (short)0, acc[0][t], false, false);
      acc[1][t] = __builtin_amdgcn_wmma_f32_16x16x32_f16(
          false, a1, false, b, (short)0, acc[1][t], false, false);
    }
  }
}

__device__ __forceinline__ float gelu_exact(float x) {
  return 0.5f * x * (1.0f + erff(x * 0.70710678118654752f));
}

// ---------------------------------------------------------------------------
// Generic NT GEMM: C[M,N] = A16[M,K] @ B16[N,K]^T (+bias)(+residual)(+gelu)
// Block = 64 threads (2 waves x 32 rows = 64 rows), grid = (N/64, M/64).
// C/D layout: lane group g = lane>>4: n = nBase + t*16 + (lane&15),
//             m = mBase + u*16 + vgpr + 8*g.
// ---------------------------------------------------------------------------
__global__ __launch_bounds__(64)
void gemm_nt_kernel(const _Float16* __restrict__ A, const _Float16* __restrict__ Bw,
                    const float* __restrict__ bias, const float* __restrict__ resid,
                    float* __restrict__ Cf, _Float16* __restrict__ Ch,
                    int M, int N, int Kd, int doGelu) {
  const int lane = threadIdx.x & 31;
  const int wave = threadIdx.x >> 5;
  const int mBase = blockIdx.y * 64 + wave * 32;
  const int nBase = blockIdx.x * 64;
  float8 acc[2][4] = {};
  wmma_nt_32x64(A + (size_t)mBase * Kd, Kd, Bw + (size_t)nBase * Kd, Kd, Kd, lane, acc);
  const int g = lane >> 4, nl = lane & 15;
#pragma unroll
  for (int u = 0; u < 2; ++u) {
#pragma unroll
    for (int t = 0; t < 4; ++t) {
      const int n = nBase + t * 16 + nl;
      const float bv = bias ? bias[n] : 0.0f;
#pragma unroll
      for (int r = 0; r < 8; ++r) {
        const int m = mBase + u * 16 + r + 8 * g;
        float v = acc[u][t][r] + bv;
        if (resid) v += resid[(size_t)m * N + n];
        if (doGelu) v = gelu_exact(v);
        const size_t o = (size_t)m * N + n;
        if (Cf) Cf[o] = v;
        if (Ch) Ch[o] = (_Float16)v;
      }
    }
  }
}

// Attention scores: per (b,h): S[256,256] = Q[256,64] @ K[256,64]^T (scale in softmax)
__global__ __launch_bounds__(64)
void attn_scores_kernel(const _Float16* __restrict__ Q, const _Float16* __restrict__ Km,
                        float* __restrict__ S) {
  const int bh = blockIdx.z;
  const int lane = threadIdx.x & 31, wave = threadIdx.x >> 5;
  const int mBase = blockIdx.y * 64 + wave * 32;
  const int nBase = blockIdx.x * 64;
  const _Float16* Qb = Q + (size_t)bh * KC * DH;
  const _Float16* Kb = Km + (size_t)bh * KC * DH;
  float* Sb = S + (size_t)bh * KC * KC;
  float8 acc[2][4] = {};
  wmma_nt_32x64(Qb + (size_t)mBase * DH, DH, Kb + (size_t)nBase * DH, DH, DH, lane, acc);
  const int g = lane >> 4, nl = lane & 15;
#pragma unroll
  for (int u = 0; u < 2; ++u) {
#pragma unroll
    for (int t = 0; t < 4; ++t) {
      const int n = nBase + t * 16 + nl;
#pragma unroll
      for (int r = 0; r < 8; ++r) {
        const int m = mBase + u * 16 + r + 8 * g;
        Sb[(size_t)m * KC + n] = acc[u][t][r];
      }
    }
  }
}

// Wave-per-row softmax over 256 cols; applies 1/sqrt(DH)=0.125; emits f16 P.
__global__ __launch_bounds__(256)
void softmax_kernel(const float* __restrict__ S, _Float16* __restrict__ P) {
  const int row  = blockIdx.x * 8 + (threadIdx.x >> 5);
  const int lane = threadIdx.x & 31;
  const float* s = S + (size_t)row * KC;
  float v[8];
  float mx = -3.4e38f;
#pragma unroll
  for (int i = 0; i < 8; ++i) { v[i] = s[lane + i * 32] * 0.125f; mx = fmaxf(mx, v[i]); }
#pragma unroll
  for (int o = 16; o > 0; o >>= 1) mx = fmaxf(mx, __shfl_xor(mx, o, 32));
  float sum = 0.0f;
#pragma unroll
  for (int i = 0; i < 8; ++i) { v[i] = expf(v[i] - mx); sum += v[i]; }
#pragma unroll
  for (int o = 16; o > 0; o >>= 1) sum += __shfl_xor(sum, o, 32);
  const float inv = 1.0f / sum;
  _Float16* p = P + (size_t)row * KC;
#pragma unroll
  for (int i = 0; i < 8; ++i) p[lane + i * 32] = (_Float16)(v[i] * inv);
}

// Attention output: per (b,h): O[256,64] = P[256,256] @ Vt[64,256]^T, f16 out
// written directly into the [2048, 512] layout at column h*64.
__global__ __launch_bounds__(64)
void attn_out_kernel(const _Float16* __restrict__ P, const _Float16* __restrict__ Vt,
                     _Float16* __restrict__ O16) {
  const int bh = blockIdx.z;
  const int b = bh >> 3, h = bh & 7;
  const int lane = threadIdx.x & 31, wave = threadIdx.x >> 5;
  const int mBase = blockIdx.y * 64 + wave * 32;
  const _Float16* Pb = P + (size_t)bh * KC * KC;
  const _Float16* Vb = Vt + (size_t)bh * DH * KC;
  _Float16* Ob = O16 + (size_t)b * KC * DM + h * DH;
  float8 acc[2][4] = {};
  wmma_nt_32x64(Pb + (size_t)mBase * KC, KC, Vb, KC, KC, lane, acc);
  const int g = lane >> 4, nl = lane & 15;
#pragma unroll
  for (int u = 0; u < 2; ++u) {
#pragma unroll
    for (int t = 0; t < 4; ++t) {
      const int n = t * 16 + nl;
#pragma unroll
      for (int r = 0; r < 8; ++r) {
        const int m = mBase + u * 16 + r + 8 * g;
        Ob[(size_t)m * DM + n] = (_Float16)acc[u][t][r];
      }
    }
  }
}

// Repack fp32 QKV [2048,1536] -> f16 Q,K [bh,256,64] and V^T [bh,64,256]
__global__ void qkv_repack_kernel(const float* __restrict__ qkv,
                                  _Float16* __restrict__ Q, _Float16* __restrict__ Km,
                                  _Float16* __restrict__ Vt) {
  const int idx = blockIdx.x * blockDim.x + threadIdx.x;
  if (idx >= NT * DM) return;
  const int t = idx >> 9, c = idx & 511;
  const int h = c >> 6, d = c & 63;
  const int b = t >> 8, kp = t & 255;
  const int bh = b * NH + h;
  const float* base = qkv + (size_t)t * (3 * DM);
  Q[((size_t)bh * KC + kp) * DH + d]  = (_Float16)base[c];
  Km[((size_t)bh * KC + kp) * DH + d] = (_Float16)base[DM + c];
  Vt[((size_t)bh * DH + d) * KC + kp] = (_Float16)base[2 * DM + c];
}

// Block-per-row LayerNorm over 512 cols; writes f32 + f16 copies.
__global__ __launch_bounds__(128)
void layernorm_kernel(const float* __restrict__ X, const float* __restrict__ gam,
                      const float* __restrict__ bet, float* __restrict__ Hf,
                      _Float16* __restrict__ Hh) {
  const int row = blockIdx.x, tid = threadIdx.x;
  __shared__ float s1[128], s2[128];
  const float* x = X + (size_t)row * DM;
  float a = 0.0f, q = 0.0f, loc[4];
#pragma unroll
  for (int i = 0; i < 4; ++i) { float v = x[tid + i * 128]; loc[i] = v; a += v; q += v * v; }
  s1[tid] = a; s2[tid] = q; __syncthreads();
  for (int st = 64; st > 0; st >>= 1) {
    if (tid < st) { s1[tid] += s1[tid + st]; s2[tid] += s2[tid + st]; }
    __syncthreads();
  }
  const float mean = s1[0] * (1.0f / DM);
  const float var  = s2[0] * (1.0f / DM) - mean * mean;
  const float inv  = rsqrtf(var + 1e-5f);
#pragma unroll
  for (int i = 0; i < 4; ++i) {
    const int c = tid + i * 128;
    const float v = (loc[i] - mean) * inv * gam[c] + bet[c];
    Hf[(size_t)row * DM + c] = v;
    Hh[(size_t)row * DM + c] = (_Float16)v;
  }
}

// h0[t,0]=v(t); h0[t,1:512]=embed_table[pos[t]] (row length 511)
__global__ void build_h_kernel(const float* __restrict__ vstate, const int* __restrict__ pos,
                               const float* __restrict__ embed,
                               float* __restrict__ Hf, _Float16* __restrict__ Hh) {
  const int idx = blockIdx.x * blockDim.x + threadIdx.x;
  if (idx >= NT * DM) return;
  const int t = idx >> 9, d = idx & 511;
  float v;
  if (d == 0) v = vstate[t];
  else        v = embed[(size_t)pos[t] * (DM - 1) + (d - 1)];
  Hf[idx] = v;
  Hh[idx] = (_Float16)v;
}

// fx[t] = h[t,:] . Wd + bd  (wave per token)
__global__ __launch_bounds__(256)
void decode_kernel(const float* __restrict__ Hf, const float* __restrict__ Wd,
                   const float* __restrict__ bd, float* __restrict__ fx) {
  const int t = blockIdx.x * 8 + (threadIdx.x >> 5);
  const int lane = threadIdx.x & 31;
  const float* h = Hf + (size_t)t * DM;
  float s = 0.0f;
#pragma unroll
  for (int i = 0; i < 16; ++i) s += h[lane + i * 32] * Wd[lane + i * 32];
#pragma unroll
  for (int o = 16; o > 0; o >>= 1) s += __shfl_xor(s, o, 32);
  if (lane == 0) fx[t] = s + bd[0];
}

// dv = v*(fx - sum_k v*fx) per batch row
__global__ __launch_bounds__(256)
void replicator_kernel(const float* __restrict__ v, const float* __restrict__ fx,
                       float* __restrict__ dv) {
  const int b = blockIdx.x, k = threadIdx.x;
  __shared__ float sh[256];
  const float vi = v[b * KC + k];
  const float fi = fx[b * KC + k];
  sh[k] = vi * fi; __syncthreads();
  for (int st = 128; st > 0; st >>= 1) {
    if (k < st) sh[k] += sh[k + st];
    __syncthreads();
  }
  dv[b * KC + k] = vi * (fi - sh[0]);
}

__global__ void axpy_kernel(const float* __restrict__ v, const float* __restrict__ kk,
                            const float* __restrict__ tptr, float c, float* __restrict__ out) {
  const int i = blockIdx.x * blockDim.x + threadIdx.x;
  if (i >= NT) return;
  const float dt = (tptr[1] - tptr[0]) * (1.0f / NSTEPS);
  out[i] = v[i] + c * dt * kk[i];
}

__global__ void rk4_update_kernel(float* __restrict__ v, const float* __restrict__ k1,
                                  const float* __restrict__ k2, const float* __restrict__ k3,
                                  const float* __restrict__ k4, const float* __restrict__ tptr) {
  const int i = blockIdx.x * blockDim.x + threadIdx.x;
  if (i >= NT) return;
  const float dt = (tptr[1] - tptr[0]) * (1.0f / NSTEPS);
  v[i] += (dt / 6.0f) * (k1[i] + 2.0f * k2[i] + 2.0f * k3[i] + k4[i]);
}

// Wave-per-row ordered stream compaction (exactly KC nonzeros per row).
__global__ void condense_kernel(const float* __restrict__ x, float* __restrict__ val,
                                int* __restrict__ pos) {
  const int b = blockIdx.x, lane = threadIdx.x;
  for (int i = lane; i < KC; i += 32) { val[b * KC + i] = 0.0f; pos[b * KC + i] = 0; }
  int count = 0;
  for (int base = 0; base < DDIM; base += 32) {
    const int col = base + lane;
    const float v = (col < DDIM) ? x[(size_t)b * DDIM + col] : 0.0f;
    unsigned long long m = __ballot(v != 0.0f) & 0xffffffffull;
    const int pre = __popcll(m & ((1ull << lane) - 1ull));
    if (v != 0.0f) {
      const int idx = count + pre;
      if (idx < KC) { val[b * KC + idx] = v; pos[b * KC + idx] = col + 1; }
    }
    count += (int)__popcll(m);
  }
}

__global__ void decondense_kernel(const float* __restrict__ val, const int* __restrict__ pos,
                                  float* __restrict__ out) {
  const int i = blockIdx.x * blockDim.x + threadIdx.x;
  if (i >= NT) return;
  const int p = pos[i];
  if (p > 0) out[(size_t)(i >> 8) * DDIM + (p - 1)] = val[i];
}

__global__ void f32_to_f16_kernel(const float* __restrict__ src, _Float16* __restrict__ dst,
                                  int n) {
  int i = blockIdx.x * blockDim.x + threadIdx.x;
  const int stride = gridDim.x * blockDim.x;
  for (; i < n; i += stride) dst[i] = (_Float16)src[i];
}

__global__ void zero_f32_kernel(float* __restrict__ p, int n) {
  int i = blockIdx.x * blockDim.x + threadIdx.x;
  const int stride = gridDim.x * blockDim.x;
  for (; i < n; i += stride) p[i] = 0.0f;
}

// ---------------------------------------------------------------------------
extern "C" void kernel_launch(void* const* d_in, const int* in_sizes, int n_in,
                              void* d_out, int out_size, void* d_ws, size_t ws_size,
                              hipStream_t stream) {
  // Inputs (setup_inputs order)
  const float* x      = (const float*)d_in[0];
  const float* tarr   = (const float*)d_in[1];
  const float* embed  = (const float*)d_in[2];
  const float* Wqkv   = (const float*)d_in[3];
  const float* bqkv   = (const float*)d_in[4];
  const float* Wo     = (const float*)d_in[5];
  const float* bo     = (const float*)d_in[6];
  const float* ln1s   = (const float*)d_in[7];
  const float* ln1b   = (const float*)d_in[8];
  const float* W1     = (const float*)d_in[9];
  const float* b1     = (const float*)d_in[10];
  const float* W2     = (const float*)d_in[11];
  const float* b2     = (const float*)d_in[12];
  const float* ln2s   = (const float*)d_in[13];
  const float* ln2b   = (const float*)d_in[14];
  const float* Wd     = (const float*)d_in[15];
  const float* bd     = (const float*)d_in[16];
  float* out = (float*)d_out;

  // Workspace bump allocator (256B aligned)
  char* wsp = (char*)d_ws;
  size_t off = 0;
  auto walloc = [&](size_t bytes) -> void* {
    off = (off + 255) & ~(size_t)255;
    void* p = wsp + off;
    off += bytes;
    return p;
  };

  _Float16* Wqkv16 = (_Float16*)walloc((size_t)NL * 3 * DM * DM * 2);
  _Float16* Wo16   = (_Float16*)walloc((size_t)NL * DM * DM * 2);
  _Float16* W116   = (_Float16*)walloc((size_t)NL * FF * DM * 2);
  _Float16* W216   = (_Float16*)walloc((size_t)NL * DM * FF * 2);
  float*    hf     = (float*)   walloc((size_t)NT * DM * 4);
  _Float16* hh     = (_Float16*)walloc((size_t)NT * DM * 2);
  float*    qkvf   = (float*)   walloc((size_t)NT * 3 * DM * 4);
  _Float16* Q16    = (_Float16*)walloc((size_t)BH * KC * DH * 2);
  _Float16* K16    = (_Float16*)walloc((size_t)BH * KC * DH * 2);
  _Float16* Vt16   = (_Float16*)walloc((size_t)BH * DH * KC * 2);
  float*    Sbuf   = (float*)   walloc((size_t)BH * KC * KC * 4);
  _Float16* P16    = (_Float16*)walloc((size_t)BH * KC * KC * 2);
  _Float16* O16    = (_Float16*)walloc((size_t)NT * DM * 2);
  _Float16* ff16   = (_Float16*)walloc((size_t)NT * FF * 2);
  float*    tmpf   = (float*)   walloc((size_t)NT * DM * 4);
  float*    fx     = (float*)   walloc((size_t)NT * 4);
  float*    vcur   = (float*)   walloc((size_t)NT * 4);
  int*      pos    = (int*)     walloc((size_t)NT * 4);
  float*    kk1    = (float*)   walloc((size_t)NT * 4);
  float*    kk2    = (float*)   walloc((size_t)NT * 4);
  float*    kk3    = (float*)   walloc((size_t)NT * 4);
  float*    kk4    = (float*)   walloc((size_t)NT * 4);
  float*    vtmp   = (float*)   walloc((size_t)NT * 4);
  (void)ws_size; (void)in_sizes; (void)n_in; (void)out_size;

  // Convert weights to f16 once per call (L2-resident afterwards)
  f32_to_f16_kernel<<<4096, 256, 0, stream>>>(Wqkv, Wqkv16, NL * 3 * DM * DM);
  f32_to_f16_kernel<<<2048, 256, 0, stream>>>(Wo,   Wo16,   NL * DM * DM);
  f32_to_f16_kernel<<<4096, 256, 0, stream>>>(W1,   W116,   NL * FF * DM);
  f32_to_f16_kernel<<<4096, 256, 0, stream>>>(W2,   W216,   NL * DM * FF);

  zero_f32_kernel<<<(NB * DDIM + 255) / 256, 256, 0, stream>>>(out, NB * DDIM);
  condense_kernel<<<NB, 32, 0, stream>>>(x, vcur, pos);

  // One derivative evaluation: vstate -> kout
  auto deriv = [&](const float* vstate, float* kout) {
    build_h_kernel<<<(NT * DM + 255) / 256, 256, 0, stream>>>(vstate, pos, embed, hf, hh);
    for (int l = 0; l < NL; ++l) {
      const _Float16* Wq_l  = Wqkv16 + (size_t)l * 3 * DM * DM;
      const _Float16* Wo_l  = Wo16   + (size_t)l * DM * DM;
      const _Float16* W1_l  = W116   + (size_t)l * FF * DM;
      const _Float16* W2_l  = W216   + (size_t)l * DM * FF;
      // QKV projection: [2048,512] x [1536,512]^T
      gemm_nt_kernel<<<dim3(3 * DM / 64, NT / 64), 64, 0, stream>>>(
          hh, Wq_l, bqkv + l * 3 * DM, nullptr, qkvf, nullptr, NT, 3 * DM, DM, 0);
      qkv_repack_kernel<<<(NT * DM + 255) / 256, 256, 0, stream>>>(qkvf, Q16, K16, Vt16);
      attn_scores_kernel<<<dim3(KC / 64, KC / 64, BH), 64, 0, stream>>>(Q16, K16, Sbuf);
      softmax_kernel<<<BH * KC / 8, 256, 0, stream>>>(Sbuf, P16);
      attn_out_kernel<<<dim3(1, KC / 64, BH), 64, 0, stream>>>(P16, Vt16, O16);
      // Output projection + residual, then LN1
      gemm_nt_kernel<<<dim3(DM / 64, NT / 64), 64, 0, stream>>>(
          O16, Wo_l, bo + l * DM, hf, tmpf, nullptr, NT, DM, DM, 0);
      layernorm_kernel<<<NT, 128, 0, stream>>>(tmpf, ln1s + l * DM, ln1b + l * DM, hf, hh);
      // FFN1 + exact GELU (f16 out)
      gemm_nt_kernel<<<dim3(FF / 64, NT / 64), 64, 0, stream>>>(
          hh, W1_l, b1 + l * FF, nullptr, nullptr, ff16, NT, FF, DM, 1);
      // FFN2 + residual, then LN2
      gemm_nt_kernel<<<dim3(DM / 64, NT / 64), 64, 0, stream>>>(
          ff16, W2_l, b2 + l * DM, hf, tmpf, nullptr, NT, DM, FF, 0);
      layernorm_kernel<<<NT, 128, 0, stream>>>(tmpf, ln2s + l * DM, ln2b + l * DM, hf, hh);
    }
    decode_kernel<<<NT / 8, 256, 0, stream>>>(hf, Wd, bd, fx);
    replicator_kernel<<<NB, 256, 0, stream>>>(vstate, fx, kout);
  };

  // RK4, 2 fixed steps
  for (int step = 0; step < NSTEPS; ++step) {
    deriv(vcur, kk1);
    axpy_kernel<<<(NT + 255) / 256, 256, 0, stream>>>(vcur, kk1, tarr, 0.5f, vtmp);
    deriv(vtmp, kk2);
    axpy_kernel<<<(NT + 255) / 256, 256, 0, stream>>>(vcur, kk2, tarr, 0.5f, vtmp);
    deriv(vtmp, kk3);
    axpy_kernel<<<(NT + 255) / 256, 256, 0, stream>>>(vcur, kk3, tarr, 1.0f, vtmp);
    deriv(vtmp, kk4);
    rk4_update_kernel<<<(NT + 255) / 256, 256, 0, stream>>>(vcur, kk1, kk2, kk3, kk4, tarr);
  }

  decondense_kernel<<<(NT + 255) / 256, 256, 0, stream>>>(vcur, pos, out);
}
